// MotionGraph_48232482734182
// MI455X (gfx1250) — matrix operand order repaired and checked by reading, not compiled, not run on hardware
//
#include <hip/hip_runtime.h>
#include <hip/hip_bf16.h>

#define NNODE 1024
#define EE    16384
#define DD    64
#define FFD   2048
#define HHEAD 8
#define HDIM  8
#define FRAMES 125

typedef __attribute__((ext_vector_type(16))) __bf16 v16bf;
typedef __attribute__((ext_vector_type(8)))  float  v8f;

__device__ __forceinline__ __bf16 f2bf(float f) {
    unsigned u = __builtin_bit_cast(unsigned, f);
    unsigned r = u + 0x7FFFu + ((u >> 16) & 1u);   // round-to-nearest-even
    unsigned short h = (unsigned short)(r >> 16);
    return __builtin_bit_cast(__bf16, h);
}
__device__ __forceinline__ float bf2f(__bf16 b) {
    unsigned short h = __builtin_bit_cast(unsigned short, b);
    unsigned u = ((unsigned)h) << 16;
    return __builtin_bit_cast(float, u);
}

// ---- fragment-layout swizzles (ISA 7.12.2) --------------------------------
// A 16x32 tile: lane<16 holds K in {0..7,16..23}, lane>=16 holds {8..15,24..31}
__device__ __forceinline__ size_t aswz(int m, int k, int Kpad) {
    int mt = m >> 4, kt = k >> 5, kin = k & 31;
    int lane = (((kin >> 3) & 1) << 4) + (m & 15);
    int j = (kin & 7) + ((kin & 16) ? 8 : 0);
    return ((size_t)(mt * (Kpad >> 5) + kt) << 9) + (lane << 4) + j;
}
// B 32x16 tile: lane<16 holds K in {0..7,16..23}, lane>=16 holds {8..15,24..31}
__device__ __forceinline__ size_t bswz(int k, int n, int Nq) {
    int kt = k >> 5, kin = k & 31, nt = n >> 4;
    int lane = (((kin >> 3) & 1) << 4) + (n & 15);
    int j = (kin & 7) + ((kin & 16) ? 8 : 0);
    return ((size_t)(kt * (Nq >> 4) + nt) << 9) + (lane << 4) + j;
}

// ---------------- static device buffers (capture-safe, no hipMalloc) -------
__device__ __align__(64) __bf16 g_ef [EE * 128];   // edge features (A-swizzled)
__device__ __align__(64) __bf16 g_m1 [EE * DD];    // edge MLP hidden (A-swizzled)
__device__ float  g_m2 [EE * DD];                  // edge messages (f32, row-major)
__device__ float  g_node[NNODE * DD];
__device__ float  g_h  [NNODE * DD];
__device__ float  g_q  [NNODE * DD], g_km[NNODE * DD], g_vm[NNODE * DD];
__device__ float  g_at [NNODE * DD], g_ao[NNODE * DD];
__device__ float  g_x1 [NNODE * DD], g_ff2[NNODE * DD], g_x2[NNODE * DD];
__device__ __align__(64) __bf16 g_hbf[NNODE * DD], g_atbf[NNODE * DD];
__device__ __align__(64) __bf16 g_x1bf[NNODE * DD], g_x2bf[NNODE * DD], g_hid[NNODE * DD];
__device__ __align__(64) __bf16 g_ffb[NNODE * FFD];
// bf16 weights, B-swizzled
__device__ __align__(64) __bf16 g_wec11[32 * DD], g_wec12[DD * DD];
__device__ __align__(64) __bf16 g_wec21[128 * DD], g_wec22[DD * DD];
__device__ __align__(64) __bf16 g_wq[DD * DD], g_wk[DD * DD], g_wv[DD * DD], g_wo[DD * DD];
__device__ __align__(64) __bf16 g_wl1[DD * FFD], g_wl2[FFD * DD], g_wd1[DD * DD];

enum { B_EF = 0, B_M1, B_HBF, B_ATBF, B_X1BF, B_X2BF, B_FFB, B_HID,
       B_WEC11, B_WEC12, B_WEC21, B_WEC22,
       B_WQ, B_WK, B_WV, B_WO, B_WL1, B_WL2, B_WD1 };
enum { F_M2 = 0, F_NODE, F_H, F_Q, F_K, F_V, F_AT, F_AO, F_X1, F_FF2, F_X2 };

__device__ __forceinline__ __bf16* bf_buf(int id) {
    switch (id) {
        case B_EF: return g_ef;       case B_M1: return g_m1;
        case B_HBF: return g_hbf;     case B_ATBF: return g_atbf;
        case B_X1BF: return g_x1bf;   case B_X2BF: return g_x2bf;
        case B_FFB: return g_ffb;     case B_HID: return g_hid;
        case B_WEC11: return g_wec11; case B_WEC12: return g_wec12;
        case B_WEC21: return g_wec21; case B_WEC22: return g_wec22;
        case B_WQ: return g_wq;       case B_WK: return g_wk;
        case B_WV: return g_wv;       case B_WO: return g_wo;
        case B_WL1: return g_wl1;     case B_WL2: return g_wl2;
        case B_WD1: return g_wd1;
    }
    return g_ef;
}
__device__ __forceinline__ float* f_buf(int id) {
    switch (id) {
        case F_M2: return g_m2;   case F_NODE: return g_node;
        case F_H: return g_h;     case F_Q: return g_q;
        case F_K: return g_km;    case F_V: return g_vm;
        case F_AT: return g_at;   case F_AO: return g_ao;
        case F_X1: return g_x1;   case F_FF2: return g_ff2;
        case F_X2: return g_x2;
    }
    return g_h;
}

// ---- f32 -> bf16 convert + swizzle. mode 0: B-frag (weight), 1: A-frag ----
__global__ void k_cvt(const float* __restrict__ src_ext, int isrc,
                      int R, int C, int ib, int Rp, int Cp, int mode) {
    int tid = blockIdx.x * 256 + threadIdx.x;
    if (tid >= Rp * Cp) return;
    int r = tid / Cp, c = tid - r * Cp;
    const float* src = (isrc >= 0) ? f_buf(isrc) : src_ext;
    float v = (r < R && c < C) ? src[(size_t)r * C + c] : 0.0f;
    size_t off = mode ? aswz(r, c, Cp) : bswz(r, c, Cp);
    bf_buf(ib)[off] = f2bf(v);
}

// ---- edge gather: ef[e] = [x_i | x_j - x_i], pad to Kpad, A-swizzled ------
__global__ void k_gather(const float* __restrict__ feat_ext, int use_h,
                         int ld, int F, int Kpad,
                         const int* __restrict__ src, const int* __restrict__ dst) {
    int tid = blockIdx.x * 256 + threadIdx.x;
    if (tid >= EE * Kpad) return;
    int e = tid / Kpad, j = tid - e * Kpad;
    const float* feat = use_h ? f_buf(F_H) : feat_ext;
    int di = dst[e], si = src[e];
    float v = 0.0f;
    if (j < F)            v = feat[(size_t)di * ld + j];
    else if (j < 2 * F) { int c = j - F;
                          v = feat[(size_t)si * ld + c] - feat[(size_t)di * ld + c]; }
    g_ef[aswz(e, j, Kpad)] = f2bf(v);
}

// ---- WMMA GEMM: 16x64 strip per wave, swizzled operands ------------------
__global__ void k_gemm(int ia, int ib, const float* __restrict__ bias,
                       int icf, int icb, int M, int Nq, int Kpad, int relu) {
    const v16bf* Asw = (const v16bf*)bf_buf(ia);
    const v16bf* Bsw = (const v16bf*)bf_buf(ib);
    int wave = threadIdx.x >> 5;
    int lane = threadIdx.x & 31;
    int ntc = Nq >> 4;           // N tiles (always mult of 4 here)
    int ntg = ntc >> 2;          // N tile groups of 4
    int t = blockIdx.x * 4 + wave;
    int tiles = (M >> 4) * ntg;
    if (t >= tiles) return;
    int mt = t / ntg, ng = t - mt * ntg;
    int ktc = Kpad >> 5;
    v8f acc[4];
#pragma unroll
    for (int i = 0; i < 4; ++i) acc[i] = (v8f){};
    for (int kt = 0; kt < ktc; ++kt) {
        v16bf a = Asw[(size_t)(mt * ktc + kt) * 32 + lane];
#pragma unroll
        for (int i = 0; i < 4; ++i) {
            v16bf b = Bsw[(size_t)(kt * ntc + (ng * 4 + i)) * 32 + lane];
            acc[i] = __builtin_amdgcn_wmma_f32_16x16x32_bf16(
                false, a, false, b, (short)0, acc[i], false, false);
        }
    }
    float* Cf = (icf >= 0) ? f_buf(icf) : nullptr;
    __bf16* Cb = (icb >= 0) ? bf_buf(icb) : nullptr;
    int mbase = (mt << 4) + ((lane >> 4) << 3);
#pragma unroll
    for (int i = 0; i < 4; ++i) {
        int ncol = ((ng * 4 + i) << 4) + (lane & 15);
        float bia = bias ? bias[ncol] : 0.0f;
#pragma unroll
        for (int r = 0; r < 8; ++r) {
            float v = acc[i][r] + bia;
            if (relu) v = fmaxf(v, 0.0f);
            int m = mbase + r;
            if (Cf) Cf[(size_t)m * Nq + ncol] = v;
            if (Cb) Cb[aswz(m, ncol, Nq)] = f2bf(v);
        }
    }
}

// ---------------- segment-max scatter --------------------------------------
__global__ void k_init_node() {
    int i = blockIdx.x * 256 + threadIdx.x;
    if (i < NNODE * DD) g_node[i] = -__builtin_inff();
}
__global__ void k_scatter(const int* __restrict__ dst) {
    int i = blockIdx.x * 256 + threadIdx.x;
    if (i >= EE * DD) return;
    int e = i >> 6, c = i & 63;
    float v = g_m2[i];
    float* p = &g_node[dst[e] * DD + c];
    if (v >= 0.0f) atomicMax((int*)p, __builtin_bit_cast(int, v));
    else           atomicMin((unsigned int*)p, __builtin_bit_cast(unsigned int, v));
}
__global__ void k_finalize() {
    int i = blockIdx.x * 256 + threadIdx.x;
    if (i >= NNODE * DD) return;
    float v = g_node[i];
    g_h[i] = (v == -__builtin_inff()) ? 0.0f : v;   // empty segment -> 0
}

// ---- attention (H=8, HD=8), online softmax, LDS K/V chunks ----------------
#define KCH 256
__global__ void k_attn() {
    __shared__ float sK[KCH][HDIM];
    __shared__ float sV[KCH][HDIM];
    int tid = threadIdx.x;
    int q = blockIdx.x * 256 + tid;
    int h = blockIdx.y;
    float qv[HDIM];
#pragma unroll
    for (int d = 0; d < HDIM; ++d) qv[d] = g_q[q * DD + h * HDIM + d];
    float m = -__builtin_inff(), l = 0.0f, o[HDIM];
#pragma unroll
    for (int d = 0; d < HDIM; ++d) o[d] = 0.0f;
    const float scale = 0.35355339059327373f;   // 1/sqrt(8)
    for (int c0 = 0; c0 < NNODE; c0 += KCH) {
        __syncthreads();
#pragma unroll
        for (int d = 0; d < HDIM; ++d) {
            sK[tid][d] = g_km[(c0 + tid) * DD + h * HDIM + d];
            sV[tid][d] = g_vm[(c0 + tid) * DD + h * HDIM + d];
        }
        __syncthreads();
        for (int kk = 0; kk < KCH; ++kk) {
            float s = 0.0f;
#pragma unroll
            for (int d = 0; d < HDIM; ++d) s += qv[d] * sK[kk][d];
            s *= scale;
            float nm = fmaxf(m, s);
            float ep = __expf(m - nm);
            float es = __expf(s - nm);
            l = l * ep + es;
#pragma unroll
            for (int d = 0; d < HDIM; ++d) o[d] = o[d] * ep + es * sV[kk][d];
            m = nm;
        }
    }
    float invl = 1.0f / l;
#pragma unroll
    for (int d = 0; d < HDIM; ++d) g_at[q * DD + h * HDIM + d] = o[d] * invl;
}

// ---- residual + LayerNorm (one wave32 per row, D=64) ----------------------
__global__ void k_ln(int ix, int ires, const float* __restrict__ gg,
                     const float* __restrict__ bb, int iof, int iob) {
    int wave = threadIdx.x >> 5, lane = threadIdx.x & 31;
    int row = blockIdx.x * 8 + wave;
    if (row >= NNODE) return;
    const float* X = f_buf(ix);
    const float* R = f_buf(ires);
    int i0 = row * DD + lane * 2;
    float z0 = X[i0] + R[i0];
    float z1 = X[i0 + 1] + R[i0 + 1];
    float s = z0 + z1;
    for (int o = 16; o; o >>= 1) s += __shfl_xor(s, o, 32);
    float mu = s * (1.0f / DD);
    float d0 = z0 - mu, d1 = z1 - mu;
    float qv = d0 * d0 + d1 * d1;
    for (int o = 16; o; o >>= 1) qv += __shfl_xor(qv, o, 32);
    float inv = rsqrtf(qv * (1.0f / DD) + 1e-5f);
    float o0 = d0 * inv * gg[lane * 2] + bb[lane * 2];
    float o1 = d1 * inv * gg[lane * 2 + 1] + bb[lane * 2 + 1];
    float* OF = f_buf(iof);
    __bf16* OB = bf_buf(iob);
    OF[i0] = o0; OF[i0 + 1] = o1;
    OB[aswz(row, lane * 2, DD)]     = f2bf(o0);
    OB[aswz(row, lane * 2 + 1, DD)] = f2bf(o1);
}

// ---- decoder layer 2 (64 -> 3), writes output frame slice -----------------
__global__ void k_dec2(int f, float* __restrict__ out,
                       const float* __restrict__ W2, const float* __restrict__ b2) {
    int n = blockIdx.x * 256 + threadIdx.x;
    if (n >= NNODE) return;
    float a0 = b2[0], a1 = b2[1], a2 = b2[2];
#pragma unroll 4
    for (int k = 0; k < DD; ++k) {
        float hv = bf2f(g_hid[aswz(n, k, DD)]);
        a0 += hv * W2[k * 3 + 0];
        a1 += hv * W2[k * 3 + 1];
        a2 += hv * W2[k * 3 + 2];
    }
    size_t base = (size_t)n * (FRAMES * 3) + (size_t)f * 3;
    out[base] = a0; out[base + 1] = a1; out[base + 2] = a2;
}

static inline int idiv(int a, int b) { return (a + b - 1) / b; }

extern "C" void kernel_launch(void* const* d_in, const int* in_sizes, int n_in,
                              void* d_out, int out_size, void* d_ws, size_t ws_size,
                              hipStream_t stream) {
    (void)in_sizes; (void)n_in; (void)out_size; (void)d_ws; (void)ws_size;
    const float* x      = (const float*)d_in[0];
    const int*   ei     = (const int*)d_in[1];
    const int*   e_src  = ei;
    const int*   e_dst  = ei + EE;
    const float* ec1_W1 = (const float*)d_in[2],  *ec1_b1 = (const float*)d_in[3];
    const float* ec1_W2 = (const float*)d_in[4],  *ec1_b2 = (const float*)d_in[5];
    const float* ec2_W1 = (const float*)d_in[6],  *ec2_b1 = (const float*)d_in[7];
    const float* ec2_W2 = (const float*)d_in[8],  *ec2_b2 = (const float*)d_in[9];
    const float* Wq = (const float*)d_in[10], *bq = (const float*)d_in[11];
    const float* Wk = (const float*)d_in[12], *bk = (const float*)d_in[13];
    const float* Wv = (const float*)d_in[14], *bv = (const float*)d_in[15];
    const float* Wo = (const float*)d_in[16], *bo = (const float*)d_in[17];
    const float* ln1_g = (const float*)d_in[18], *ln1_b = (const float*)d_in[19];
    const float* lin1_W = (const float*)d_in[20], *lin1_b = (const float*)d_in[21];
    const float* lin2_W = (const float*)d_in[22], *lin2_b = (const float*)d_in[23];
    const float* ln2_g = (const float*)d_in[24], *ln2_b = (const float*)d_in[25];
    const float* dec_W1 = (const float*)d_in[26], *dec_b1 = (const float*)d_in[27];
    const float* dec_W2 = (const float*)d_in[28], *dec_b2 = (const float*)d_in[29];
    float* out = (float*)d_out;

    auto cvt = [&](const float* src, int isrc, int R, int C, int ib,
                   int Rp, int Cp, int mode) {
        k_cvt<<<idiv(Rp * Cp, 256), 256, 0, stream>>>(src, isrc, R, C, ib, Rp, Cp, mode);
    };
    auto gemm = [&](int ia, int ib, const float* bias, int icf, int icb,
                    int M, int Nq, int Kp, int relu) {
        int tiles = (M / 16) * ((Nq / 16) / 4);
        k_gemm<<<idiv(tiles, 4), 128, 0, stream>>>(ia, ib, bias, icf, icb, M, Nq, Kp, relu);
    };
    auto agg = [&]() {
        k_init_node<<<idiv(NNODE * DD, 256), 256, 0, stream>>>();
        k_scatter<<<idiv(EE * DD, 256), 256, 0, stream>>>(e_dst);
        k_finalize<<<idiv(NNODE * DD, 256), 256, 0, stream>>>();
    };
    auto econv = [&](const float* featExt, int useH, int ld, int F, int Kp,
                     int wb1, const float* b1, int wb2, const float* b2) {
        k_gather<<<idiv(EE * Kp, 256), 256, 0, stream>>>(featExt, useH, ld, F, Kp, e_src, e_dst);
        gemm(B_EF, wb1, b1, -1, B_M1, EE, DD, Kp, 1);
        gemm(B_M1, wb2, b2, F_M2, -1, EE, DD, DD, 0);
        agg();
    };

    // ---- convert weights to bf16 (zero-padded K, B-fragment swizzle) ----
    cvt(ec1_W1, -1, 6, DD, B_WEC11, 32, DD, 0);
    cvt(ec1_W2, -1, DD, DD, B_WEC12, DD, DD, 0);
    cvt(ec2_W1, -1, 2 * DD, DD, B_WEC21, 128, DD, 0);
    cvt(ec2_W2, -1, DD, DD, B_WEC22, DD, DD, 0);
    cvt(Wq, -1, DD, DD, B_WQ, DD, DD, 0);
    cvt(Wk, -1, DD, DD, B_WK, DD, DD, 0);
    cvt(Wv, -1, DD, DD, B_WV, DD, DD, 0);
    cvt(Wo, -1, DD, DD, B_WO, DD, DD, 0);
    cvt(lin1_W, -1, DD, FFD, B_WL1, DD, FFD, 0);
    cvt(lin2_W, -1, FFD, DD, B_WL2, FFD, DD, 0);
    cvt(dec_W1, -1, DD, DD, B_WD1, DD, DD, 0);

    // ---- EdgeConv1 on x[N,3] -> g_h ----
    econv(x, 0, 3, 3, 32, B_WEC11, ec1_b1, B_WEC12, ec1_b2);
    // ---- EdgeConv2 on g_h[N,64] -> g_h ----
    econv(nullptr, 1, DD, DD, 128, B_WEC21, ec2_b1, B_WEC22, ec2_b2);

    // ---- transformer encoder layer ----
    cvt(nullptr, F_H, NNODE, DD, B_HBF, NNODE, DD, 1);
    gemm(B_HBF, B_WQ, bq, F_Q, -1, NNODE, DD, DD, 0);
    gemm(B_HBF, B_WK, bk, F_K, -1, NNODE, DD, DD, 0);
    gemm(B_HBF, B_WV, bv, F_V, -1, NNODE, DD, DD, 0);
    k_attn<<<dim3(NNODE / 256, HHEAD), 256, 0, stream>>>();
    cvt(nullptr, F_AT, NNODE, DD, B_ATBF, NNODE, DD, 1);
    gemm(B_ATBF, B_WO, bo, F_AO, -1, NNODE, DD, DD, 0);
    k_ln<<<NNODE / 8, 256, 0, stream>>>(F_H, F_AO, ln1_g, ln1_b, F_X1, B_X1BF);
    gemm(B_X1BF, B_WL1, lin1_b, -1, B_FFB, NNODE, FFD, DD, 1);
    gemm(B_FFB, B_WL2, lin2_b, F_FF2, -1, NNODE, DD, FFD, 0);
    k_ln<<<NNODE / 8, 256, 0, stream>>>(F_X1, F_FF2, ln2_g, ln2_b, F_X2, B_X2BF);

    // ---- autoregressive frame loop ----
    for (int f = 0; f < FRAMES; ++f) {
        int curA = (f == 0) ? B_X2BF : B_HBF;
        gemm(curA, B_WD1, dec_b1, -1, B_HID, NNODE, DD, DD, 1);   // decoder L1 + relu
        k_dec2<<<idiv(NNODE, 256), 256, 0, stream>>>(f, out, dec_W2, dec_b2);
        // next state: EdgeConv1 on this frame's output (read from d_out, ld=375)
        econv(out + (size_t)f * 3, 0, FRAMES * 3, 3, 32,
              B_WEC11, ec1_b1, B_WEC12, ec1_b2);
        cvt(nullptr, F_H, NNODE, DD, B_HBF, NNODE, DD, 1);
    }
}
